// HybridMemoryMultiFocalPercentCluster_44504451121873
// MI455X (gfx1250) — matrix operand order repaired and checked by reading, not compiled, not run on hardware
//
#include <hip/hip_runtime.h>

// out[B,M] = inputs[B,D] @ features[M,D]^T  (B=1024, M=100000, D=256, fp32 in/out)
// fp32 is matrix-throughput-bound on MI455X (needs 2.4 PFLOP/s to hit the 22us
// HBM floor). Use bf16x3 split (hi*hi + hi*lo + lo*hi, fp32 accum) on
// V_WMMA_F32_16X16X32_BF16: ~fp32 accuracy at bf16-XDL rate -> back to
// bandwidth-bound. Split pairs precomputed once into d_ws.

#define B_ROWS 1024
#define M_COLS 100000
#define D_K    256
#define BM 128
#define BN 128
#define LSTR 264   // bf16 per LDS row: 132 dwords == 4 (mod 64) -> perfect bank tiling

typedef float  v8f   __attribute__((ext_vector_type(8)));
typedef __bf16 v16bf __attribute__((ext_vector_type(16)));
typedef __bf16 v8bf  __attribute__((ext_vector_type(8)));
typedef __bf16 v4bf  __attribute__((ext_vector_type(4)));

static __device__ __forceinline__ v16bf concat8(v8bf a, v8bf b) {
    return __builtin_shufflevector(a, b, 0,1,2,3,4,5,6,7,8,9,10,11,12,13,14,15);
}

// ---- One-time fp32 -> (bf16 hi, bf16 lo) split, elementwise ----
__global__ __launch_bounds__(256) void split_bf16_kernel(
    const float* __restrict__ src,
    unsigned short* __restrict__ hi, unsigned short* __restrict__ lo, int n4)
{
    int idx = blockIdx.x * blockDim.x + threadIdx.x;
    if (idx >= n4) return;
    float4 v = reinterpret_cast<const float4*>(src)[idx];
    float xs[4] = {v.x, v.y, v.z, v.w};
    v4bf h, l;
    #pragma unroll
    for (int e = 0; e < 4; ++e) {
        float  x  = xs[e];
        __bf16 hh = (__bf16)x;          // RNE
        float  r  = x - (float)hh;      // exact in fp32
        h[e] = hh;
        l[e] = (__bf16)r;
    }
    reinterpret_cast<v4bf*>(hi)[idx] = h;
    reinterpret_cast<v4bf*>(lo)[idx] = l;
}

__global__ __launch_bounds__(256) void hm_forward_gemm_bf16x3_kernel(
    const unsigned short* __restrict__ Ihi, const unsigned short* __restrict__ Ilo,
    const unsigned short* __restrict__ Fhi, const unsigned short* __restrict__ Flo,
    float* __restrict__ out)
{
    extern __shared__ char smem_raw[];
    __bf16* Ahi = reinterpret_cast<__bf16*>(smem_raw);
    __bf16* Alo = Ahi + BM * LSTR;
    __bf16* Bhi = Alo + BM * LSTR;
    __bf16* Blo = Bhi + BN * LSTR;

    const int tid  = threadIdx.x;
    const int lane = tid & 31;
    const int wave = tid >> 5;           // 0..7
    const int half = lane >> 4;
    const int n16  = lane & 15;

    const int rb = blockIdx.y * BM;
    const int cb = blockIdx.x * BN;

    // ---- Stage A (inputs) hi/lo: K-contiguous, pure b128 copy, coalesced ----
    #pragma unroll
    for (int j = 0; j < 16; ++j) {
        int g   = tid + j * 256;         // 8-elem chunk id, 0..4095
        int row = g >> 5;                // 32 chunks per 256-K row
        int kc  = (g & 31) << 3;
        size_t goff = ((size_t)(rb + row) << 8) + kc;     // *D_K
        *reinterpret_cast<v8bf*>(Ahi + row * LSTR + kc) =
            *reinterpret_cast<const v8bf*>(Ihi + goff);
        *reinterpret_cast<v8bf*>(Alo + row * LSTR + kc) =
            *reinterpret_cast<const v8bf*>(Ilo + goff);
    }
    // ---- Stage B (features) hi/lo: native orientation, no transpose ----
    #pragma unroll
    for (int j = 0; j < 16; ++j) {
        int g   = tid + j * 256;
        int col = g >> 5;
        int kc  = (g & 31) << 3;
        int m   = cb + col;
        if (m > M_COLS - 1) m = M_COLS - 1;   // clamp; predicated at store
        size_t goff = ((size_t)m << 8) + kc;
        *reinterpret_cast<v8bf*>(Bhi + col * LSTR + kc) =
            *reinterpret_cast<const v8bf*>(Fhi + goff);
        *reinterpret_cast<v8bf*>(Blo + col * LSTR + kc) =
            *reinterpret_cast<const v8bf*>(Flo + goff);
    }
    __syncthreads();

    // ---- 8 waves as 4x2; wave tile = 32 rows x 64 cols = 2x4 WMMA tiles ----
    const int wr = wave & 3;
    const int wc = wave >> 2;
    const int arow0 = wr * 32;
    const int bcol0 = wc * 64;

    // Fragment base pointers (contiguous K order k = 16h + s, same for A and B):
    //  A lane(h, m=n16): K-run  [k0+16h, +16)
    //  B lane(h, n=n16): K-run  [k0+16h, +16)
    const v8bf* pAhi[2]; const v8bf* pAlo[2];
    #pragma unroll
    for (int ri = 0; ri < 2; ++ri) {
        int off = (arow0 + ri * 16 + n16) * LSTR + 16 * half;
        pAhi[ri] = reinterpret_cast<const v8bf*>(Ahi + off);
        pAlo[ri] = reinterpret_cast<const v8bf*>(Alo + off);
    }
    const v8bf* pBhi[4]; const v8bf* pBlo[4];
    #pragma unroll
    for (int ci = 0; ci < 4; ++ci) {
        int off = (bcol0 + ci * 16 + n16) * LSTR + 16 * half;
        pBhi[ci] = reinterpret_cast<const v8bf*>(Bhi + off);
        pBlo[ci] = reinterpret_cast<const v8bf*>(Blo + off);
    }

    v8f acc[2][4] = {};

    #pragma unroll 2
    for (int ks = 0; ks < 8; ++ks) {     // k0 = 32*ks; all offsets are immediates
        v16bf ahi[2], alo[2];
        #pragma unroll
        for (int ri = 0; ri < 2; ++ri) {
            ahi[ri] = concat8(pAhi[ri][4 * ks], pAhi[ri][4 * ks + 1]);
            alo[ri] = concat8(pAlo[ri][4 * ks], pAlo[ri][4 * ks + 1]);
        }
        #pragma unroll
        for (int ci = 0; ci < 4; ++ci) {
            v16bf bhi = concat8(pBhi[ci][4 * ks], pBhi[ci][4 * ks + 1]);
            v16bf blo = concat8(pBlo[ci][4 * ks], pBlo[ci][4 * ks + 1]);
            #pragma unroll
            for (int ri = 0; ri < 2; ++ri) {
                acc[ri][ci] = __builtin_amdgcn_wmma_f32_16x16x32_bf16(
                    false, ahi[ri], false, bhi, (short)0, acc[ri][ci], false, false);
                acc[ri][ci] = __builtin_amdgcn_wmma_f32_16x16x32_bf16(
                    false, ahi[ri], false, blo, (short)0, acc[ri][ci], false, false);
                acc[ri][ci] = __builtin_amdgcn_wmma_f32_16x16x32_bf16(
                    false, alo[ri], false, bhi, (short)0, acc[ri][ci], false, false);
            }
        }
    }

    // ---- Store: C/D VGPR j = row M=j+8h, col N=lane&15; non-temporal stream ----
    #pragma unroll
    for (int ci = 0; ci < 4; ++ci) {
        int col = cb + bcol0 + ci * 16 + n16;
        if (col < M_COLS) {
            #pragma unroll
            for (int ri = 0; ri < 2; ++ri) {
                int rowbase = rb + arow0 + ri * 16 + 8 * half;
                #pragma unroll
                for (int j = 0; j < 8; ++j) {
                    __builtin_nontemporal_store(
                        acc[ri][ci][j],
                        out + (size_t)(rowbase + j) * M_COLS + col);
                }
            }
        }
    }
}

extern "C" void kernel_launch(void* const* d_in, const int* in_sizes, int n_in,
                              void* d_out, int out_size, void* d_ws, size_t ws_size,
                              hipStream_t stream) {
    const float* inputs   = (const float*)d_in[0];   // [1024, 256]
    const float* features = (const float*)d_in[1];   // [100000, 256]
    float* out = (float*)d_out;                      // [1024, 100000]

    const size_t featElems = (size_t)M_COLS * D_K;   // 25,600,000
    const size_t inElems   = (size_t)B_ROWS * D_K;   //    262,144
    unsigned short* Fhi = (unsigned short*)d_ws;     // ws usage: ~103.5 MB
    unsigned short* Flo = Fhi + featElems;
    unsigned short* Ihi = Flo + featElems;
    unsigned short* Ilo = Ihi + inElems;

    // One-time (per call) fp32 -> bf16 hi/lo split.
    split_bf16_kernel<<<(int)(featElems / 4 / 256), 256, 0, stream>>>(
        features, Fhi, Flo, (int)(featElems / 4));
    split_bf16_kernel<<<(int)(inElems / 4 / 256), 256, 0, stream>>>(
        inputs, Ihi, Ilo, (int)(inElems / 4));

    dim3 block(256);
    dim3 grid((M_COLS + BN - 1) / BN, B_ROWS / BM);  // (782, 8)
    size_t smem_bytes = (size_t)4 * BM * LSTR * sizeof(__bf16); // 270,336 B

    hm_forward_gemm_bf16x3_kernel<<<grid, block, smem_bytes, stream>>>(
        Ihi, Ilo, Fhi, Flo, out);
}
